// Electronic_embedding_88622355185701
// MI455X (gfx1250) — matrix operand order repaired and checked by reading, not compiled, not run on hardware
//
#include <hip/hip_runtime.h>
#include <stdint.h>
#include <math.h>

#define FDIM 128

typedef __bf16 bf16_t;
typedef bf16_t v16bf __attribute__((ext_vector_type(16)));
typedef bf16_t v2bf  __attribute__((ext_vector_type(2)));
typedef float  v8f   __attribute__((ext_vector_type(8)));

typedef uint32_t u32x4 __attribute__((ext_vector_type(4)));
typedef int      i32x8 __attribute__((ext_vector_type(8)));
typedef int      i32x4 __attribute__((ext_vector_type(4)));

#ifdef __has_builtin
#if __has_builtin(__builtin_amdgcn_tensor_load_to_lds)
#define USE_TDM 1
#endif
#if __has_builtin(__builtin_amdgcn_s_wait_tensorcnt)
#define HAVE_TENSORCNT 1
#endif
#endif

union ABfrag { v16bf v; uint32_t u[8]; uint4 q[2]; };
union PK2 { v2bf v; uint32_t u; };
union PK1 { bf16_t v; unsigned short u; };

// native f32 -> bf16 converts (v_cvt_pk_bf16_f32 class ops, not bit-twiddling)
__device__ __forceinline__ uint32_t pk2bf(float a, float b) {
    PK2 p; p.v[0] = (bf16_t)a; p.v[1] = (bf16_t)b; return p.u;
}
__device__ __forceinline__ unsigned short f2bf(float a) {
    PK1 p; p.v = (bf16_t)a; return p.u;
}
__device__ __forceinline__ float softplus_f(float x) {
    return fmaxf(x, 0.f) + log1pf(__expf(-fabsf(x)));
}
__device__ __forceinline__ float silu_f(float x) {
    return x / (1.f + __expf(-x));
}

// ---------------------------------------------------------------------------
// TDM load of a 128x128 bf16 weight matrix (32 KB) from global into LDS.
// D# per CDNA5 ISA: 2-D tensor, data_size=2B, type=2.
// Issued by wave 0 only (uniform branch); TENSORcnt tracks completion.
// This toolchain exposes the 6-arg builtin: (g0, g1, g2, g3, g4, cpol).
// ---------------------------------------------------------------------------
__device__ __forceinline__ void load_weights_lds(const unsigned short* __restrict__ gsrc,
                                                 unsigned short* lds_dst, int tid, int wave) {
#ifdef USE_TDM
    if (wave == 0) {
        uint64_t ga = (uint64_t)(uintptr_t)gsrc;
        uint32_t lo = (uint32_t)(uintptr_t)lds_dst;   // flat addr low 32 bits == LDS byte offset
        u32x4 g0;
        g0[0] = 1u;                                   // count = 1 (valid descriptor)
        g0[1] = lo;                                   // lds_addr
        g0[2] = (uint32_t)ga;                         // global_addr[31:0]
        g0[3] = ((uint32_t)(ga >> 32) & 0x01FFFFFFu)  // global_addr[56:32]
                | (2u << 30);                         // type = 2 ("image")
        i32x8 g1;
        g1[0] = 0x00010000;                           // data_size = 1 -> 2 bytes
        g1[1] = (int)(128u << 16);                    // tensor_dim0 = 128
        g1[2] = (int)(128u << 16);                    // tensor_dim1 = 128
        g1[3] = (int)(128u << 16);                    // tile_dim0   = 128
        g1[4] = 128;                                  // tile_dim1   = 128, tile_dim2 = 0
        g1[5] = 128;                                  // tensor_dim0_stride = 128
        g1[6] = 0;
        g1[7] = 0;
        i32x4 g2 = {0, 0, 0, 0};
        i32x4 g3 = {0, 0, 0, 0};
        i32x8 g4 = {0, 0, 0, 0, 0, 0, 0, 0};
        __builtin_amdgcn_tensor_load_to_lds(g0, g1, g2, g3, g4, 0);
#ifdef HAVE_TENSORCNT
        __builtin_amdgcn_s_wait_tensorcnt(0);
#endif
    }
#else
    const uint4* src = (const uint4*)gsrc;
    uint4* dst = (uint4*)lds_dst;
    for (int i = tid; i < FDIM * FDIM / 8; i += 256) dst[i] = src[i];
#endif
    (void)tid; (void)wave;
}

// ---------------------------------------------------------------------------
// Prep: k_batch[b,f] = |E[b]|*Wk[f] + bk[f]; bf16-transpose Wq/W1/W2; zero sums
// ---------------------------------------------------------------------------
__global__ __launch_bounds__(256) void prep_kernel(
    const float* __restrict__ E,  const float* __restrict__ Wk, const float* __restrict__ bk,
    const float* __restrict__ Wq, const float* __restrict__ W1, const float* __restrict__ W2,
    float* __restrict__ kbatch, float* __restrict__ pred, float* __restrict__ natom,
    unsigned short* __restrict__ WqT, unsigned short* __restrict__ W1T,
    unsigned short* __restrict__ W2T, int B) {
    int idx = blockIdx.x * blockDim.x + threadIdx.x;
    if (idx < B * FDIM) {
        int b = idx >> 7, f = idx & 127;
        kbatch[idx] = fabsf(E[b]) * Wk[f] + bk[f];
    }
    if (idx < FDIM * FDIM) {
        int k = idx >> 7, n = idx & 127;
        WqT[n * FDIM + k] = f2bf(Wq[idx]);   // store transposed, row-major [n][k]
        W1T[n * FDIM + k] = f2bf(W1[idx]);
        W2T[n * FDIM + k] = f2bf(W2[idx]);
    }
    if (idx < B) { pred[idx] = 0.f; natom[idx] = 0.f; }
}

// Build B-fragment (32x16 bf16) from LDS transposed weights.
// lane<16: N = n0+lane, K = kk..kk+7,kk+16..kk+23 ; lane>=16: N = n0+lane-16, K = kk+8..kk+15,kk+24..kk+31
__device__ __forceinline__ void load_bfrag(ABfrag& b, const unsigned short* sW,
                                           int n0, int kk, int l, int half) {
    const unsigned short* p = &sW[(n0 + l) * FDIM + kk + half * 8];
    b.q[0] = *(const uint4*)p;
    b.q[1] = *(const uint4*)(p + 16);
}

// ---------------------------------------------------------------------------
// Phase 1: q = x@Wq + bq ; embed = softplus(k*q) -> d_out ; segment sums
// One wave = 16 atom rows; 8 waves/block = 128 rows/block.
// ---------------------------------------------------------------------------
__global__ __launch_bounds__(256) void phase1_kernel(
    const float* __restrict__ x, const int* __restrict__ seg,
    const float* __restrict__ bq, const float* __restrict__ kbatch,
    const unsigned short* __restrict__ WqT,
    float* __restrict__ embed_out, float* __restrict__ pred,
    float* __restrict__ natom, int N) {
    __shared__ unsigned short sW[FDIM * FDIM];
    const int wave = threadIdx.x >> 5, lane = threadIdx.x & 31;
    const int half = lane >> 4, l = lane & 15;
    const int base = blockIdx.x * 128 + wave * 16;

    load_weights_lds(WqT, sW, threadIdx.x, wave);
    __syncthreads();

    v8f c[8];
#pragma unroll
    for (int t = 0; t < 8; ++t)
#pragma unroll
        for (int r = 0; r < 8; ++r) c[t][r] = 0.f;

    // A rows come straight from global x (row-major, contiguous 32B per group)
    const int arow = min(base + l, N - 1);
    const float* px = x + (size_t)arow * FDIM;

#pragma unroll
    for (int ks = 0; ks < 4; ++ks) {
        const int kk = ks * 32;
        ABfrag a;
        const int k0 = kk + half * 8;
#pragma unroll
        for (int j = 0; j < 4; ++j) a.u[j]     = pk2bf(px[k0 + 2 * j],      px[k0 + 2 * j + 1]);
#pragma unroll
        for (int j = 0; j < 4; ++j) a.u[4 + j] = pk2bf(px[k0 + 16 + 2 * j], px[k0 + 17 + 2 * j]);
#pragma unroll
        for (int t = 0; t < 8; ++t) {
            ABfrag b;
            load_bfrag(b, sW, t * 16, kk, l, half);
            c[t] = __builtin_amdgcn_wmma_f32_16x16x32_bf16(
                false, a.v, false, b.v, (short)0, c[t], false, false);
        }
    }

    // Epilogue: C layout -> VGPR r holds row (r + half*8), col = t*16 + l
    int segs[8]; float rowsum[8];
#pragma unroll
    for (int r = 0; r < 8; ++r) {
        int row = base + half * 8 + r;
        segs[r] = seg[min(row, N - 1)];
        rowsum[r] = 0.f;
    }
#pragma unroll
    for (int t = 0; t < 8; ++t) {
        const int col = t * 16 + l;
        const float bias = bq[col];
#pragma unroll
        for (int r = 0; r < 8; ++r) {
            int row = base + half * 8 + r;
            float qv = c[t][r] + bias;
            float kv = kbatch[segs[r] * FDIM + col];
            float e = softplus_f(kv * qv);
            rowsum[r] += e;
            if (row < N) embed_out[(size_t)row * FDIM + col] = e;
        }
    }
#pragma unroll
    for (int r = 0; r < 8; ++r) {
        float s = rowsum[r];
        s += __shfl_xor(s, 1, 32);
        s += __shfl_xor(s, 2, 32);
        s += __shfl_xor(s, 4, 32);
        s += __shfl_xor(s, 8, 32);   // stays within 16-lane half
        int row = base + half * 8 + r;
        if (l == 0 && row < N) {
            atomicAdd(&pred[segs[r]], s);
            atomicAdd(&natom[segs[r]], 1.f);
        }
    }
}

// ---------------------------------------------------------------------------
// Phase 2: scale = embed + d/F ; out = scale + W2(silu(W1(silu(scale))+b1))+b2
// embed is read from d_out, final result overwrites d_out (block-disjoint rows).
// ---------------------------------------------------------------------------
__global__ __launch_bounds__(256) void phase2_kernel(
    const float* __restrict__ E, const int* __restrict__ seg,
    const float* __restrict__ pred, const float* __restrict__ natom,
    const float* __restrict__ b1, const float* __restrict__ b2,
    const unsigned short* __restrict__ W1T, const unsigned short* __restrict__ W2T,
    float* __restrict__ out, int N) {
    __shared__ unsigned short sW[FDIM * FDIM];        // 32 KB
    __shared__ unsigned short sStage[8 * 16 * FDIM];  // 32 KB, per-wave 16x128 bf16
    const int wave = threadIdx.x >> 5, lane = threadIdx.x & 31;
    const int half = lane >> 4, l = lane & 15;
    const int base = blockIdx.x * 128 + wave * 16;
    unsigned short* myStage = &sStage[wave * 16 * FDIM];

    load_weights_lds(W1T, sW, threadIdx.x, wave);

    // per-row correction d = (E - pred)/natom / F
    float dval[8];
#pragma unroll
    for (int r = 0; r < 8; ++r) {
        int rc = min(base + half * 8 + r, N - 1);
        int sg = seg[rc];
        dval[r] = (E[sg] - pred[sg]) / natom[sg] * (1.0f / FDIM);
    }

    // read embed, build scale (kept in f32 regs, C layout), stage silu(scale) bf16
    v8f s[8];
#pragma unroll
    for (int t = 0; t < 8; ++t) {
        const int col = t * 16 + l;
#pragma unroll
        for (int r = 0; r < 8; ++r) {
            int rc = min(base + half * 8 + r, N - 1);
            float sc = out[(size_t)rc * FDIM + col] + dval[r];
            s[t][r] = sc;
            myStage[(half * 8 + r) * FDIM + col] = f2bf(silu_f(sc));
        }
    }
    __syncthreads();

    // GEMM1: h = silu(scale) @ W1
    v8f h[8];
#pragma unroll
    for (int t = 0; t < 8; ++t)
#pragma unroll
        for (int r = 0; r < 8; ++r) h[t][r] = 0.f;
#pragma unroll
    for (int ks = 0; ks < 4; ++ks) {
        const int kk = ks * 32;
        ABfrag a;
        a.q[0] = *(const uint4*)&myStage[l * FDIM + kk + half * 8];
        a.q[1] = *(const uint4*)&myStage[l * FDIM + kk + 16 + half * 8];
#pragma unroll
        for (int t = 0; t < 8; ++t) {
            ABfrag b;
            load_bfrag(b, sW, t * 16, kk, l, half);
            h[t] = __builtin_amdgcn_wmma_f32_16x16x32_bf16(
                false, a.v, false, b.v, (short)0, h[t], false, false);
        }
    }

    // bias + silu, restage
#pragma unroll
    for (int t = 0; t < 8; ++t) {
        const int col = t * 16 + l;
        const float bb = b1[col];
#pragma unroll
        for (int r = 0; r < 8; ++r)
            myStage[(half * 8 + r) * FDIM + col] = f2bf(silu_f(h[t][r] + bb));
    }
    __syncthreads();   // all waves done with W1 in sW
    load_weights_lds(W2T, sW, threadIdx.x, wave);
    __syncthreads();

    // GEMM2
#pragma unroll
    for (int t = 0; t < 8; ++t)
#pragma unroll
        for (int r = 0; r < 8; ++r) h[t][r] = 0.f;
#pragma unroll
    for (int ks = 0; ks < 4; ++ks) {
        const int kk = ks * 32;
        ABfrag a;
        a.q[0] = *(const uint4*)&myStage[l * FDIM + kk + half * 8];
        a.q[1] = *(const uint4*)&myStage[l * FDIM + kk + 16 + half * 8];
#pragma unroll
        for (int t = 0; t < 8; ++t) {
            ABfrag b;
            load_bfrag(b, sW, t * 16, kk, l, half);
            h[t] = __builtin_amdgcn_wmma_f32_16x16x32_bf16(
                false, a.v, false, b.v, (short)0, h[t], false, false);
        }
    }

    // out = scale + h + b2
#pragma unroll
    for (int t = 0; t < 8; ++t) {
        const int col = t * 16 + l;
        const float bb = b2[col];
#pragma unroll
        for (int r = 0; r < 8; ++r) {
            int row = base + half * 8 + r;
            if (row < N) out[(size_t)row * FDIM + col] = s[t][r] + h[t][r] + bb;
        }
    }
}

// ---------------------------------------------------------------------------
extern "C" void kernel_launch(void* const* d_in, const int* in_sizes, int n_in,
                              void* d_out, int out_size, void* d_ws, size_t ws_size,
                              hipStream_t stream) {
    const float* x  = (const float*)d_in[0];
    const float* E  = (const float*)d_in[1];
    // d_in[2] = num_batch scalar; use array sizes instead
    const int*   sg = (const int*)d_in[3];
    const float* Wq = (const float*)d_in[4];
    const float* bq = (const float*)d_in[5];
    const float* Wk = (const float*)d_in[6];
    const float* bk = (const float*)d_in[7];
    const float* W1 = (const float*)d_in[8];
    const float* b1 = (const float*)d_in[9];
    const float* W2 = (const float*)d_in[10];
    const float* b2 = (const float*)d_in[11];
    float* out = (float*)d_out;

    const int N = in_sizes[0] / FDIM;   // 500000
    const int B = in_sizes[1];          // 4096

    // workspace carve (~2.2 MB)
    char* w = (char*)d_ws;
    float* kbatch = (float*)w;           w += (size_t)B * FDIM * sizeof(float);
    float* pred   = (float*)w;           w += (size_t)B * sizeof(float);
    float* natom  = (float*)w;           w += (size_t)B * sizeof(float);
    unsigned short* WqT = (unsigned short*)w; w += FDIM * FDIM * sizeof(unsigned short);
    unsigned short* W1T = (unsigned short*)w; w += FDIM * FDIM * sizeof(unsigned short);
    unsigned short* W2T = (unsigned short*)w; w += FDIM * FDIM * sizeof(unsigned short);

    prep_kernel<<<(B * FDIM + 255) / 256, 256, 0, stream>>>(
        E, Wk, bk, Wq, W1, W2, kbatch, pred, natom, WqT, W1T, W2T, B);

    const int nblk = (N + 127) / 128;
    phase1_kernel<<<nblk, 256, 0, stream>>>(
        x, sg, bq, kbatch, WqT, out, pred, natom, N);
    phase2_kernel<<<nblk, 256, 0, stream>>>(
        E, sg, pred, natom, b1, b2, W1T, W2T, out, N);
}
